// MWEConv_6519760355566
// MI455X (gfx1250) — matrix operand order repaired and checked by reading, not compiled, not run on hardware
//
#include <hip/hip_runtime.h>
#include <hip/hip_bf16.h>

#define N_NODES   50000
#define N_EDGES   800000
#define FEATS     128
#define NCHAN     4
#define PITCH     136          // LDS row pitch in halves (keeps rows 16B aligned)
#define ROWT      5            // row tiles (of 16) per block
#define BROWS     80           // 50000 = 625 * 80, exact

typedef __attribute__((ext_vector_type(16))) __bf16         v16bf;
typedef __attribute__((ext_vector_type(8)))  float          v8f;
typedef __attribute__((ext_vector_type(8)))  unsigned short ushort8;

union Frag {
    v16bf   v;
    ushort8 h[2];
};

// round-to-nearest-even f32 -> bf16 (bit trick)
__device__ __forceinline__ unsigned int f2bf(float x) {
    unsigned int u = __float_as_uint(x);
    u += 0x7FFFu + ((u >> 16) & 1u);
    return u >> 16;
}
__device__ __forceinline__ unsigned int pack_bf16(float lo, float hi) {
    return f2bf(lo) | (f2bf(hi) << 16);
}

// ---------------------------------------------------------------------------
// Kernel 1: zero the aggregation workspace (4 * 50000 * 128 f32)
// ---------------------------------------------------------------------------
__global__ __launch_bounds__(256) void zero_ws(float4* __restrict__ p, long n4) {
    long i      = (long)blockIdx.x * blockDim.x + threadIdx.x;
    long stride = (long)gridDim.x * blockDim.x;
    float4 z = {0.f, 0.f, 0.f, 0.f};
    for (; i < n4; i += stride) p[i] = z;
}

// ---------------------------------------------------------------------------
// Kernel 2: edge-weighted scatter-add. One wave32 per edge; each lane owns
// 4 consecutive features (float4). HW f32 atomics into L2.
// ---------------------------------------------------------------------------
__global__ __launch_bounds__(256) void edge_scatter(
    const float* __restrict__ node_state,   // [N, 128]
    const float* __restrict__ edge_w,       // [E, 4]
    const int*   __restrict__ src,
    const int*   __restrict__ dst,
    float*       __restrict__ agg)          // [C][N][128]
{
    const int e    = (int)((blockIdx.x * 256u + threadIdx.x) >> 5);
    const int lane = threadIdx.x & 31;
    if (e >= N_EDGES) return;

    const int s = src[e];
    const int d = dst[e];
    const float4 x = ((const float4*)(node_state + (size_t)s * FEATS))[lane];
    const float4 w = ((const float4*)edge_w)[e];

    const size_t cstride = (size_t)N_NODES * FEATS;
    float* dbase = agg + (size_t)d * FEATS + lane * 4;

    const float wc[4] = {w.x, w.y, w.z, w.w};
#pragma unroll
    for (int c = 0; c < NCHAN; ++c) {
        float* p = dbase + (size_t)c * cstride;
        unsafeAtomicAdd(p + 0, wc[c] * x.x);
        unsafeAtomicAdd(p + 1, wc[c] * x.y);
        unsafeAtomicAdd(p + 2, wc[c] * x.z);
        unsafeAtomicAdd(p + 3, wc[c] * x.w);
    }
}

// ---------------------------------------------------------------------------
// Kernel 3: fused per-channel GEMM + ReLU + channel sum + final GEMM.
// Block = 256 threads (8 waves); block owns 80 node rows (5 row-tiles).
// Wave w owns output columns [w*16, w*16+16). K = 128 via
// v_wmma_f32_16x16x32_bf16; each B fragment is reused for 5 A fragments.
// ---------------------------------------------------------------------------
__global__ __launch_bounds__(256) void fused_gemm(
    const float* __restrict__ agg,      // [C][N][128] f32
    const float* __restrict__ weight,   // [128][128][4] f32  (k, n, c)
    const float* __restrict__ bias,     // [128][4] f32       (n, c)
    const float* __restrict__ final_w,  // [128][128] f32     (n, k)
    const float* __restrict__ final_b,  // [128] f32
    float*       __restrict__ out)      // [N][128] f32
{
    extern __shared__ __align__(16) unsigned short smem[];
    unsigned short* Wt = smem;                    // [128][PITCH]  B operand (n-major)
    unsigned short* Xt = Wt + 128 * PITCH;        // [80][PITCH]   A operand
    unsigned short* Ht = Xt + BROWS * PITCH;      // [80][PITCH]   h tile (bf16)

    const int tid  = threadIdx.x;
    const int wave = tid >> 5;          // 0..7  -> n-tile
    const int lane = tid & 31;
    const int rh   = lane >> 4;         // 0/1 half of the wave
    const int l16  = lane & 15;
    const int row0 = blockIdx.x * BROWS;

    const int ncol = wave * 16 + l16;   // this lane's global output column

    v8f hacc[ROWT];
#pragma unroll
    for (int rt = 0; rt < ROWT; ++rt)
        hacc[rt] = (v8f){0.f, 0.f, 0.f, 0.f, 0.f, 0.f, 0.f, 0.f};

    for (int c = 0; c < NCHAN; ++c) {
        __syncthreads();   // protect Wt/Xt reuse from previous iteration

        // Stage W_c transposed: Wt[n][k] = weight[k][n][c], packed bf16 pairs
        for (int p = tid; p < 8192; p += 256) {
            int n = p >> 6, k = (p & 63) * 2;
            float lo = weight[((size_t)k * 128 + n) * 4 + c];
            float hi = weight[((size_t)(k + 1) * 128 + n) * 4 + c];
            *(unsigned int*)&Wt[n * PITCH + k] = pack_bf16(lo, hi);
        }
        // Stage aggregated-feature tile: Xt[m][k] = agg[c][row0+m][k]
        const float* ag = agg + ((size_t)c * N_NODES + row0) * FEATS;
        for (int p = tid; p < BROWS * 64; p += 256) {
            int m = p >> 6, k = (p & 63) * 2;
            float2 v = *(const float2*)(ag + (size_t)m * FEATS + k);
            *(unsigned int*)&Xt[m * PITCH + k] = pack_bf16(v.x, v.y);
        }
        __syncthreads();

        v8f acc[ROWT];
#pragma unroll
        for (int rt = 0; rt < ROWT; ++rt)
            acc[rt] = (v8f){0.f, 0.f, 0.f, 0.f, 0.f, 0.f, 0.f, 0.f};

#pragma unroll
        for (int k0 = 0; k0 < 128; k0 += 32) {
            Frag B;   // col N = ncol, K = k0 + rh*8 + (0..7), then +16; reused 5x
            B.h[0] = *(const ushort8*)&Wt[ncol * PITCH + k0 + rh * 8];
            B.h[1] = *(const ushort8*)&Wt[ncol * PITCH + k0 + 16 + rh * 8];
#pragma unroll
            for (int rt = 0; rt < ROWT; ++rt) {
                Frag A;   // row M = rt*16 + l16, K = k0 + rh*8 + (0..7), then +16
                const int mrow = rt * 16 + l16;
                A.h[0] = *(const ushort8*)&Xt[mrow * PITCH + k0 + rh * 8];
                A.h[1] = *(const ushort8*)&Xt[mrow * PITCH + k0 + 16 + rh * 8];
                acc[rt] = __builtin_amdgcn_wmma_f32_16x16x32_bf16(
                    false, A.v, false, B.v, (short)0, acc[rt], false, false);
            }
        }

        const float bv = bias[ncol * 4 + c];
#pragma unroll
        for (int rt = 0; rt < ROWT; ++rt)
#pragma unroll
            for (int r = 0; r < 8; ++r) {
                float t = acc[rt][r] + bv;
                hacc[rt][r] += (t > 0.f) ? t : 0.f;   // ReLU then channel-sum
            }
    }

    // h tile -> LDS (bf16). C/D layout: VGPR r <-> row (rt*16 + r + 8*rh)
#pragma unroll
    for (int rt = 0; rt < ROWT; ++rt)
#pragma unroll
        for (int r = 0; r < 8; ++r)
            Ht[(rt * 16 + r + 8 * rh) * PITCH + ncol] = (unsigned short)f2bf(hacc[rt][r]);

    __syncthreads();   // Ht complete; all phase-1 reads of Wt done

    // Stage final_w: out = h @ final_w^T -> B[k][n] = final_w[n][k],
    // i.e. n-major rows exactly as stored (contiguous float2 reads).
    for (int p = tid; p < 8192; p += 256) {
        int n = p >> 6, k = (p & 63) * 2;
        float2 v = *(const float2*)(final_w + (size_t)n * FEATS + k);
        *(unsigned int*)&Wt[n * PITCH + k] = pack_bf16(v.x, v.y);
    }
    __syncthreads();

    v8f acc[ROWT];
#pragma unroll
    for (int rt = 0; rt < ROWT; ++rt)
        acc[rt] = (v8f){0.f, 0.f, 0.f, 0.f, 0.f, 0.f, 0.f, 0.f};

#pragma unroll
    for (int k0 = 0; k0 < 128; k0 += 32) {
        Frag B;
        B.h[0] = *(const ushort8*)&Wt[ncol * PITCH + k0 + rh * 8];
        B.h[1] = *(const ushort8*)&Wt[ncol * PITCH + k0 + 16 + rh * 8];
#pragma unroll
        for (int rt = 0; rt < ROWT; ++rt) {
            Frag A;
            const int mrow = rt * 16 + l16;
            A.h[0] = *(const ushort8*)&Ht[mrow * PITCH + k0 + rh * 8];
            A.h[1] = *(const ushort8*)&Ht[mrow * PITCH + k0 + 16 + rh * 8];
            acc[rt] = __builtin_amdgcn_wmma_f32_16x16x32_bf16(
                false, A.v, false, B.v, (short)0, acc[rt], false, false);
        }
    }

    const float fb = final_b[ncol];
#pragma unroll
    for (int rt = 0; rt < ROWT; ++rt)
#pragma unroll
        for (int r = 0; r < 8; ++r) {
            int m = row0 + rt * 16 + r + 8 * rh;
            out[(size_t)m * FEATS + ncol] = acc[rt][r] + fb;
        }
}

// ---------------------------------------------------------------------------
extern "C" void kernel_launch(void* const* d_in, const int* in_sizes, int n_in,
                              void* d_out, int out_size, void* d_ws, size_t ws_size,
                              hipStream_t stream) {
    const float* node_state = (const float*)d_in[0];   // [50000,128]
    const float* edge_w     = (const float*)d_in[1];   // [800000,4]
    const float* weight     = (const float*)d_in[2];   // [128,128,4]
    const float* bias       = (const float*)d_in[3];   // [128,4]
    const float* final_w    = (const float*)d_in[4];   // [128,128]
    const float* final_b    = (const float*)d_in[5];   // [128]
    const int*   src        = (const int*)d_in[6];     // [800000]
    const int*   dst        = (const int*)d_in[7];     // [800000]
    float*       out        = (float*)d_out;           // [50000,128]

    float* agg = (float*)d_ws;   // [4][50000][128] f32 = 102.4 MB

    const long n4 = (long)NCHAN * N_NODES * FEATS / 4;
    zero_ws<<<4096, 256, 0, stream>>>((float4*)agg, n4);

    const int edge_blocks = (N_EDGES + 7) / 8;         // 8 waves/block
    edge_scatter<<<edge_blocks, 256, 0, stream>>>(node_state, edge_w, src, dst, agg);

    const int smem_bytes = (128 * PITCH + 2 * BROWS * PITCH) * (int)sizeof(unsigned short);
    fused_gemm<<<N_NODES / BROWS, 256, smem_bytes, stream>>>(
        agg, weight, bias, final_w, final_b, out);
}
